// DisentangledMultiheadAttention_59648505806966
// MI455X (gfx1250) — matrix operand order repaired and checked by reading, not compiled, not run on hardware
//
#include <hip/hip_runtime.h>

// ---------------------------------------------------------------------------
// DeBERTa disentangled attention on MI455X (gfx1250, wave32, WMMA bf16 + TDM).
// ---------------------------------------------------------------------------
#define B_      8
#define S_      384
#define IN_DIM_ 512
#define E_      512
#define H_      8
#define D_      64
#define NREL    1023          // 2*MAXL-1 rows of rel_table
#define NRELP   1024          // padded row count in workspace
#define SCALE_  0.07216878364870323f   // 1/sqrt(64*3)

typedef __attribute__((ext_vector_type(16))) __bf16 v16bf;
typedef __attribute__((ext_vector_type(8)))  float  v8f;
typedef __attribute__((ext_vector_type(4)))  unsigned tdm_g0_t;
typedef __attribute__((ext_vector_type(8)))  unsigned tdm_g1_t;

// ---------------- bf16 helpers (bit-level, RNE) ----------------------------
static __device__ __forceinline__ unsigned short f32_to_bf16(float f) {
  union { float f; unsigned u; } x; x.f = f;
  unsigned u = x.u;
  u += 0x7FFFu + ((u >> 16) & 1u);       // round-to-nearest-even
  return (unsigned short)(u >> 16);
}

// ---------------- WMMA wrapper ---------------------------------------------
static __device__ __forceinline__ v8f wmma_bf16(v16bf a, v16bf b, v8f c) {
  return __builtin_amdgcn_wmma_f32_16x16x32_bf16(
      /*neg_a=*/false, a, /*neg_b=*/false, b,
      /*c_mod=*/(short)0, c, /*reuse_a=*/false, /*reuse_b=*/false);
}

// ---------------- TDM: 2-D tile load, Global -> LDS ------------------------
// D# per CDNA5 ISA:
//   group0: count=1 | lds_addr | global_addr | type=2
//   group1: data_size=1 (2B) | tensor_dim0/1 | tile_dim0/1 | dim0_stride
// 2-D tensor -> VADDR2/VADDR3 omitted (NULL).  Tracked by TENSORcnt.
static __device__ __forceinline__ void
tdm_load_2d_bf16(const void* gptr, unsigned lds_byte_addr,
                 unsigned dim0_elems, unsigned dim1_rows,
                 unsigned tile0_elems, unsigned tile1_rows,
                 unsigned dim0_stride_elems) {
  unsigned long long ga = (unsigned long long)gptr;
  tdm_g0_t g0;
  g0[0] = 1u;                                  // count=1
  g0[1] = lds_byte_addr;                       // lds_addr
  g0[2] = (unsigned)ga;                        // global_addr[31:0]
  g0[3] = (unsigned)(ga >> 32) | (2u << 30);   // global_addr[56:32] | type=2
  tdm_g1_t g1;
  g1[0] = (1u << 16);                          // wg_mask=0 | data_size=1 (2B)
  g1[1] = (dim0_elems & 0xFFFFu) << 16;        // tensor_dim0[15:0]
  g1[2] = (dim0_elems >> 16) | ((dim1_rows & 0xFFFFu) << 16);
  g1[3] = (dim1_rows >> 16) | (tile0_elems << 16);   // tile_dim0
  g1[4] = tile1_rows & 0xFFFFu;                // tile_dim1 | tile_dim2=0
  g1[5] = dim0_stride_elems;                   // tensor_dim0_stride[31:0]
  g1[6] = 0u;                                  // stride0 hi | stride1 lo
  g1[7] = 0u;                                  // stride1 hi
  asm volatile("tensor_load_to_lds %0, %1" :: "s"(g0), "s"(g1) : "memory");
}

// ---------------- fragment loaders (contiguous k order, same for A and B) ------------------
// A: 16x32 bf16. lane<16: row M=lane,    K = 0..15
//                lane>=16: row M=lane-16, K = 16..31
static __device__ __forceinline__ v16bf
load_a_frag(const unsigned short* src, int ld, int row0, int k0, int lane) {
  int m     = row0 + (lane & 15);
  int kbase = (lane < 16) ? 0 : 16;
  const __bf16* p = (const __bf16*)(src + (size_t)m * ld + k0 + kbase);
  v16bf a;
#pragma unroll
  for (int e = 0; e < 8; ++e) a[e] = p[e];
#pragma unroll
  for (int e = 0; e < 8; ++e) a[8 + e] = p[8 + e];
  return a;
}

// B: 32x16 bf16 with B[k][n] = W[n0+n][k0+k]  (W row-major, stride ld)
// lane<16: N=lane, K=0..15 ; lane>=16: N=lane-16, K=16..31  (K contiguous)
static __device__ __forceinline__ v16bf
load_b_frag(const unsigned short* src, int ld, int n0, int k0, int lane) {
  int n  = n0 + (lane & 15);
  int kb = (lane < 16) ? 0 : 16;
  const __bf16* p = (const __bf16*)(src + (size_t)n * ld + k0 + kb);
  v16bf v;
#pragma unroll
  for (int e = 0; e < 16; ++e) v[e] = p[e];
  return v;
}

// B: 32x16 bf16 with B[k][n] = M[k0+k][n0+n]  (M row-major, stride ld; K strided)
static __device__ __forceinline__ v16bf
load_b_kstride(const unsigned short* src, int ld, int n0, int k0, int lane) {
  int n  = n0 + (lane & 15);
  int kb = (lane < 16) ? 0 : 16;
  const unsigned short* p = src + (size_t)(k0 + kb) * ld + n;
  v16bf v;
#pragma unroll
  for (int e = 0; e < 16; ++e) v[e] = *(const __bf16*)(p + (size_t)e * ld);
  return v;
}

// ---------------- kernel: f32 -> bf16 convert ------------------------------
__global__ void cvt_bf16_kernel(const float* __restrict__ src,
                                unsigned short* __restrict__ dst, int n) {
  for (int i = blockIdx.x * blockDim.x + threadIdx.x; i < n;
       i += gridDim.x * blockDim.x)
    dst[i] = f32_to_bf16(src[i]);
}

// ---------------- kernel: bf16 WMMA GEMM (16x64 strip / wave) --------------
// C[M,N] = A[M,K] @ W[N,K]^T + bias[N]   (A,W bf16 row-major; C f32 or bf16)
// A-strip (16 x K) staged into LDS by the Tensor Data Mover; one A fragment
// feeds 4 WMMAs per k-step (16x64 output strip per wave).
template <bool OUT_F32>
__global__ __launch_bounds__(32)
void gemm_bf16_kernel(const unsigned short* __restrict__ A,
                      const unsigned short* __restrict__ W,
                      const float* __restrict__ bias,
                      void* __restrict__ Cout,
                      int M, int N, int K) {
  __shared__ unsigned short Atile[16 * IN_DIM_];   // 16 KB (K <= 512)
  int lane = threadIdx.x & 31;
  int n0   = blockIdx.x * 64;
  int m0   = blockIdx.y * 16;

  tdm_load_2d_bf16(A + (size_t)m0 * K, (unsigned)(size_t)&Atile[0],
                   (unsigned)K, 16u, (unsigned)K, 16u, (unsigned)K);
  __builtin_amdgcn_s_wait_tensorcnt(0);

  v8f c0 = {}, c1 = {}, c2 = {}, c3 = {};
  for (int kk = 0; kk < K; kk += 32) {
    v16bf a  = load_a_frag(Atile, K, 0, kk, lane);
    v16bf b0 = load_b_frag(W, K, n0,      kk, lane);
    v16bf b1 = load_b_frag(W, K, n0 + 16, kk, lane);
    v16bf b2 = load_b_frag(W, K, n0 + 32, kk, lane);
    v16bf b3 = load_b_frag(W, K, n0 + 48, kk, lane);
    c0 = wmma_bf16(a, b0, c0);
    c1 = wmma_bf16(a, b1, c1);
    c2 = wmma_bf16(a, b2, c2);
    c3 = wmma_bf16(a, b3, c3);
  }

  int mh = (lane < 16) ? 0 : 8;
  int nl = lane & 15;
#pragma unroll
  for (int nt = 0; nt < 4; ++nt) {
    const v8f& c = (nt == 0) ? c0 : (nt == 1) ? c1 : (nt == 2) ? c2 : c3;
    int   n  = n0 + nt * 16 + nl;
    float bv = bias[n];
#pragma unroll
    for (int r = 0; r < 8; ++r) {
      int m = m0 + r + mh;
      if (m < M) {
        float v = c[r] + bv;
        if (OUT_F32) ((float*)Cout)[(size_t)m * N + n] = v;
        else ((unsigned short*)Cout)[(size_t)m * N + n] = f32_to_bf16(v);
      }
    }
  }
}

// ---------------- kernel: disentangled attention ---------------------------
// One wave per (b, h, 16-row query tile).  Full 16xS logit strip in LDS,
// c2c via WMMA, banded c2p/p2c via WMMA + LDS gather, softmax, P@V via WMMA.
__global__ __launch_bounds__(32)
void attn_kernel(const unsigned short* __restrict__ Qb,
                 const unsigned short* __restrict__ Kb,
                 const unsigned short* __restrict__ Vb,
                 const unsigned short* __restrict__ PKb,
                 const unsigned short* __restrict__ PQb,
                 const int* __restrict__ mask,
                 unsigned short* __restrict__ valsb) {
  __shared__ float          lg[16][S_];       // logits strip (f32)
  __shared__ float          c2pb[16][32];     // c2p band  [u][r']
  __shared__ float          p2cb[16][32];     // p2c band  [w][r']
  __shared__ unsigned short pr[16][S_];       // unnormalized probs (bf16)
  __shared__ float          rsum[16];

  int lane = threadIdx.x & 31;
  const int ntile = S_ / 16;
  int it = blockIdx.x % ntile;
  int h  = (blockIdx.x / ntile) % H_;
  int b  = blockIdx.x / (ntile * H_);
  int i0 = it * 16;

  const unsigned short* Qbase = Qb + ((size_t)b * S_ + i0) * E_ + h * D_;
  const unsigned short* Kall  = Kb + (size_t)b * S_ * E_ + h * D_;
  const unsigned short* Vall  = Vb + (size_t)b * S_ * E_ + h * D_;

  for (int j0 = 0; j0 < S_; j0 += 16) {
    const unsigned short* Kbase = Kall + (size_t)j0 * E_;
    int rbase = j0 - i0 + 496;                 // = (j0-i0) + 511 - 15, in [128,864]

    if (j0 + 16 < S_)                          // prefetch next K tile (L2 hint)
      __builtin_prefetch(Kall + (size_t)(j0 + 16) * E_ + lane * E_, 0, 0);

    v8f acc_cc = {}, acc_cp0 = {}, acc_cp1 = {}, acc_pc0 = {}, acc_pc1 = {};
#pragma unroll
    for (int ks = 0; ks < D_; ks += 32) {
      v16bf aq = load_a_frag(Qbase, E_, 0, ks, lane);          // Q rows i0..i0+15
      v16bf ak = load_a_frag(Kbase, E_, 0, ks, lane);          // K rows j0..j0+15
      v16bf bk = load_b_frag(Kbase, E_, 0, ks, lane);          // K^T
      acc_cc = wmma_bf16(aq, bk, acc_cc);
      v16bf bp0 = load_b_frag(PKb + (size_t)rbase * E_ + h * D_, E_, 0,  ks, lane);
      v16bf bp1 = load_b_frag(PKb + (size_t)rbase * E_ + h * D_, E_, 16, ks, lane);
      acc_cp0 = wmma_bf16(aq, bp0, acc_cp0);
      acc_cp1 = wmma_bf16(aq, bp1, acc_cp1);
      v16bf bq0 = load_b_frag(PQb + (size_t)rbase * E_ + h * D_, E_, 0,  ks, lane);
      v16bf bq1 = load_b_frag(PQb + (size_t)rbase * E_ + h * D_, E_, 16, ks, lane);
      acc_pc0 = wmma_bf16(ak, bq0, acc_pc0);
      acc_pc1 = wmma_bf16(ak, bq1, acc_pc1);
    }

    __syncthreads();                           // prior band reads done
    int rowh = (lane >> 4) << 3;               // 0 or 8
    int col  = lane & 15;
#pragma unroll
    for (int r = 0; r < 8; ++r) {
      int u = r + rowh;
      c2pb[u][col]      = acc_cp0[r];
      c2pb[u][col + 16] = acc_cp1[r];
      p2cb[u][col]      = acc_pc0[r];          // 'u' is K-tile row = w here
      p2cb[u][col + 16] = acc_pc1[r];
    }
    __syncthreads();

#pragma unroll
    for (int r = 0; r < 8; ++r) {
      int u  = r + rowh;
      int w  = col;
      int rp = w - u + 15;                     // in [0,30]
      float l = (acc_cc[r] + c2pb[u][rp] + p2cb[w][rp]) * SCALE_;
      if (mask[(size_t)b * S_ + j0 + w] == 0) l = -9.0e15f;
      lg[u][j0 + w] = l;
    }
  }
  __syncthreads();

  // ---- softmax: lane handles row (lane&15), half (lane>>4); shfl-combine --
  {
    int u  = lane & 15;
    int jb = (lane >> 4) * (S_ / 2);
    float mx = -3.0e38f;
    for (int j = 0; j < S_ / 2; ++j) mx = fmaxf(mx, lg[u][jb + j]);
    mx = fmaxf(mx, __shfl_xor(mx, 16, 32));
    float sum = 0.f;
    for (int j = 0; j < S_ / 2; ++j) {
      float e = __expf(lg[u][jb + j] - mx);
      sum += e;
      pr[u][jb + j] = f32_to_bf16(e);          // unnormalized; /sum in epilogue
    }
    sum += __shfl_xor(sum, 16, 32);
    if (lane < 16) rsum[u] = sum;
  }
  __syncthreads();

  // ---- P @ V via WMMA, normalize in epilogue ------------------------------
  int rowh = (lane >> 4) << 3;
  int col  = lane & 15;
#pragma unroll
  for (int nt = 0; nt < 4; ++nt) {
    v8f acc = {};
    for (int k0 = 0; k0 < S_; k0 += 32) {
      v16bf ap = load_a_frag(&pr[0][0], S_, 0, k0, lane);
      v16bf bv = load_b_kstride(Vall + nt * 16, E_, 0, k0, lane);
      acc = wmma_bf16(ap, bv, acc);
    }
#pragma unroll
    for (int r = 0; r < 8; ++r) {
      int u = r + rowh;
      float v = acc[r] / rsum[u];
      valsb[((size_t)b * S_ + i0 + u) * E_ + h * D_ + nt * 16 + col] =
          f32_to_bf16(v);
    }
  }
}

// ---------------------------------------------------------------------------
extern "C" void kernel_launch(void* const* d_in, const int* in_sizes, int n_in,
                              void* d_out, int out_size, void* d_ws,
                              size_t ws_size, hipStream_t stream) {
  (void)in_sizes; (void)n_in; (void)out_size; (void)ws_size;

  const float* x    = (const float*)d_in[0];
  const int*   mask = (const int*)d_in[1];
  const float* Wq   = (const float*)d_in[2];
  const float* bq   = (const float*)d_in[3];
  const float* Wk   = (const float*)d_in[4];
  const float* bk   = (const float*)d_in[5];
  const float* Wv   = (const float*)d_in[6];
  const float* bv   = (const float*)d_in[7];
  const float* rel  = (const float*)d_in[8];
  const float* Wpk  = (const float*)d_in[9];
  const float* bpk  = (const float*)d_in[10];
  const float* Wpq  = (const float*)d_in[11];
  const float* bpq  = (const float*)d_in[12];
  const float* Wo   = (const float*)d_in[13];
  const float* bo   = (const float*)d_in[14];
  float* out = (float*)d_out;

  // ---- carve bf16 staging buffers out of d_ws ----
  char*  ws  = (char*)d_ws;
  size_t off = 0;
  auto carve = [&](size_t elems) {
    unsigned short* p = (unsigned short*)(ws + off);
    off = (off + elems * sizeof(unsigned short) + 255) & ~(size_t)255;
    return p;
  };
  const size_t MS = (size_t)B_ * S_;           // 3072 token rows
  unsigned short* xb   = carve(MS * IN_DIM_);
  unsigned short* Wqb  = carve((size_t)E_ * IN_DIM_);
  unsigned short* Wkb  = carve((size_t)E_ * IN_DIM_);
  unsigned short* Wvb  = carve((size_t)E_ * IN_DIM_);
  unsigned short* Wpkb = carve((size_t)E_ * IN_DIM_);
  unsigned short* Wpqb = carve((size_t)E_ * IN_DIM_);
  unsigned short* Wob  = carve((size_t)IN_DIM_ * E_);
  unsigned short* relb = carve((size_t)NRELP * IN_DIM_);
  unsigned short* Qb   = carve(MS * E_);
  unsigned short* Kb   = carve(MS * E_);
  unsigned short* Vb   = carve(MS * E_);
  unsigned short* PKb  = carve((size_t)NRELP * E_);
  unsigned short* PQb  = carve((size_t)NRELP * E_);
  unsigned short* Valb = carve(MS * E_);

  // ---- 1) f32 -> bf16 converts ----
  auto cvt = [&](const float* s, unsigned short* d, int n) {
    int blk = (n + 255) / 256; if (blk > 2048) blk = 2048;
    cvt_bf16_kernel<<<blk, 256, 0, stream>>>(s, d, n);
  };
  cvt(x,   xb,   (int)(MS * IN_DIM_));
  cvt(Wq,  Wqb,  E_ * IN_DIM_);
  cvt(Wk,  Wkb,  E_ * IN_DIM_);
  cvt(Wv,  Wvb,  E_ * IN_DIM_);
  cvt(Wpk, Wpkb, E_ * IN_DIM_);
  cvt(Wpq, Wpqb, E_ * IN_DIM_);
  cvt(Wo,  Wob,  IN_DIM_ * E_);
  cvt(rel, relb, NREL * IN_DIM_);

  // ---- 2) projection GEMMs (bf16 out), A-strips staged via TDM ----
  dim3 blk32(32, 1, 1);
  {
    dim3 g(E_ / 64, (int)(MS / 16));
    gemm_bf16_kernel<false><<<g, blk32, 0, stream>>>(xb, Wqb, bq, Qb, (int)MS, E_, IN_DIM_);
    gemm_bf16_kernel<false><<<g, blk32, 0, stream>>>(xb, Wkb, bk, Kb, (int)MS, E_, IN_DIM_);
    gemm_bf16_kernel<false><<<g, blk32, 0, stream>>>(xb, Wvb, bv, Vb, (int)MS, E_, IN_DIM_);
  }
  {
    dim3 g(E_ / 64, (NREL + 15) / 16);
    gemm_bf16_kernel<false><<<g, blk32, 0, stream>>>(relb, Wpkb, bpk, PKb, NREL, E_, IN_DIM_);
    gemm_bf16_kernel<false><<<g, blk32, 0, stream>>>(relb, Wpqb, bpq, PQb, NREL, E_, IN_DIM_);
  }

  // ---- 3) disentangled attention ----
  {
    dim3 g(B_ * H_ * (S_ / 16));
    attn_kernel<<<g, blk32, 0, stream>>>(Qb, Kb, Vb, PKb, PQb, mask, Valb);
  }

  // ---- 4) output projection (f32 out) ----
  {
    dim3 g(IN_DIM_ / 64, (int)(MS / 16));
    gemm_bf16_kernel<true><<<g, blk32, 0, stream>>>(Valb, Wob, bo, out, (int)MS, IN_DIM_, E_);
  }
}
